// MambaVisionMixer_29334626632151
// MI455X (gfx1250) — matrix operand order repaired and checked by reading, not compiled, not run on hardware
//
#include <hip/hip_runtime.h>
#include <hip/hip_bf16.h>
#include <math.h>

typedef __attribute__((ext_vector_type(16))) _Float16 v16h;
typedef __attribute__((ext_vector_type(8)))  _Float16 v8h;
typedef __attribute__((ext_vector_type(8)))  float    v8f;

#define BSZ   8
#define LSEQ  2048
#define MROWS (BSZ * LSEQ)        // 16384
#define DMODEL 1024
#define DINNER 2048
#define DHALF  1024
#define DSTATE 16
#define DTRANK 64
#define XDBL_N (DTRANK + 2 * DSTATE)  // 96

// LDS B-panel: 64 n-rows x 32 k halves, padded row pitch 40 halves (80B)
#define BP_PITCH 40
#define BP_HALVES (64 * BP_PITCH)     // one buffer, in halves

__device__ __forceinline__ float softplus_f(float v) {
    return (v > 20.0f) ? v : log1pf(__expf(v));
}
__device__ __forceinline__ float silu_f(float v) {
    return v * (1.0f / (1.0f + __expf(-v)));
}

// ---------------------------------------------------------------------------
// Convert f32 -> f16 (flat)
// ---------------------------------------------------------------------------
__global__ void cvt16_kernel(const float* __restrict__ in,
                             _Float16* __restrict__ out, int n) {
    int i = blockIdx.x * blockDim.x + threadIdx.x;
    if (i < n) out[i] = (_Float16)in[i];
}

// ---------------------------------------------------------------------------
// Convert + transpose weights: W[K][N] f32 -> WT[N][K] f16
// ---------------------------------------------------------------------------
__global__ void cvtT16_kernel(const float* __restrict__ W,
                              _Float16* __restrict__ WT, int K, int N) {
    int i = blockIdx.x * blockDim.x + threadIdx.x;
    if (i >= K * N) return;
    int k = i / N, n = i - k * N;          // coalesced read
    WT[(size_t)n * K + k] = (_Float16)W[i];
}

// ---------------------------------------------------------------------------
// WMMA GEMM: C[M,N] = act(A16[M,K] @ BT16[N,K]^T + bias[N])
// A row-major f16 via global_load_b128; B (pre-transposed [N][K]) staged
// cooperatively into LDS with async global->LDS b128 copies (ASYNCcnt),
// double-buffered so the DMA for step k+1 overlaps the WMMAs of step k.
// 4 waves/block; each wave: MT(=2) 16-row tiles x NT(=4) 16-col tiles
// -> 8 WMMA per 32-deep k-step; B fragments shared by all 4 waves via LDS.
// Requires M%128==0, K%32==0. GUARD=true adds a uniform n-bound check
// (only needed for the N=96 GEMM).
// ---------------------------------------------------------------------------
template <int ACT, bool STORE16, bool GUARD>
__global__ void wmma_gemm16_kernel(const _Float16* __restrict__ A,
                                   const _Float16* __restrict__ BT,
                                   float* __restrict__ C,
                                   _Float16* __restrict__ C16,
                                   const float* __restrict__ bias,
                                   int M, int N, int K,
                                   int lda, int ldc, int ldc16) {
    constexpr int NT = 4;
    constexpr int MT = 2;
    __shared__ _Float16 bsm[2 * BP_HALVES];   // 10 KB, assumed at LDS offset 0

    const int tid  = threadIdx.x;
    const int lane = tid & 31;
    const int wave = tid >> 5;                // 0..3
    const int half = lane >> 4;               // 0 or 1
    const int mr   = lane & 15;

    const int m0     = (blockIdx.x * 4 + wave) * (16 * MT);
    const int n0base = blockIdx.y * (16 * NT);

    v8f acc[MT][NT] = {};

    const _Float16* arow[MT];
#pragma unroll
    for (int mt = 0; mt < MT; ++mt) {
        arow[mt] = A + (size_t)(m0 + mt * 16 + mr) * lda + half * 8;
    }

    // ---- async B-panel stage: 64 rows x 64B, 256 x 16B chunks, 128 thr x2 ----
    const int srow0 = tid >> 2;               // 0..31
    const int schnk = tid & 3;                // 0..3 (16B chunks within 64B row)
    const _Float16* bbase = BT + (size_t)(n0base + srow0) * K + schnk * 8;
    auto stage = [&](int k0, int buf) {
#pragma unroll
        for (int it = 0; it < 2; ++it) {
            const int r = srow0 + it * 32;
            const _Float16* gsrc = bbase + (size_t)it * 32 * K + k0;
            const unsigned lds_b =
                (unsigned)((buf * BP_HALVES + r * BP_PITCH + schnk * 8) * 2);
            asm volatile("global_load_async_to_lds_b128 %0, %1, off"
                         :: "v"(lds_b), "v"(gsrc) : "memory");
        }
    };

    stage(0, 0);
    asm volatile("s_wait_asynccnt 0x0" ::: "memory");
    __syncthreads();

    int cur = 0;
    for (int k0 = 0; k0 < K; k0 += 32) {
        if (k0 + 32 < K) stage(k0 + 32, cur ^ 1);

        // A fragments: a[e] = A[m][k0 + (e/8)*16 + half*8 + (e%8)]
        v16h afrag[MT];
#pragma unroll
        for (int mt = 0; mt < MT; ++mt) {
            v8h alo = *(const v8h*)(arow[mt] + k0);
            v8h ahi = *(const v8h*)(arow[mt] + k0 + 16);
            afrag[mt] = __builtin_shufflevector(alo, ahi,
                0, 1, 2, 3, 4, 5, 6, 7, 8, 9, 10, 11, 12, 13, 14, 15);
            __builtin_prefetch(arow[mt] + k0 + 64, 0, 1);  // global_prefetch_b8
        }
#pragma unroll
        for (int t = 0; t < NT; ++t) {
            if (GUARD && (n0base + t * 16 >= N)) break;    // uniform
            // B fragment from LDS: b[e] = BT[n0+mr][k0+(e/8)*16+half*8+(e%8)]
            const _Float16* bp =
                &bsm[cur * BP_HALVES + (t * 16 + mr) * BP_PITCH + half * 8];
            v8h blo = *(const v8h*)bp;
            v8h bhi = *(const v8h*)(bp + 16);
            v16h bfrag = __builtin_shufflevector(blo, bhi,
                0, 1, 2, 3, 4, 5, 6, 7, 8, 9, 10, 11, 12, 13, 14, 15);
#pragma unroll
            for (int mt = 0; mt < MT; ++mt) {
                acc[mt][t] = __builtin_amdgcn_wmma_f32_16x16x32_f16(
                    false, afrag[mt], false, bfrag, (short)0,
                    acc[mt][t], false, false);
            }
        }

        asm volatile("s_wait_asynccnt 0x0" ::: "memory");
        __syncthreads();
        cur ^= 1;
    }

    // store: n = n0 + mr, m = m0 + mt*16 + r + half*8
#pragma unroll
    for (int t = 0; t < NT; ++t) {
        const int n0 = n0base + t * 16;
        if (GUARD && n0 >= N) break;
        const int n  = n0 + mr;
        const float bv = (bias != nullptr) ? bias[n] : 0.0f;
#pragma unroll
        for (int mt = 0; mt < MT; ++mt) {
#pragma unroll
            for (int r = 0; r < 8; ++r) {
                const int m = m0 + mt * 16 + r + half * 8;
                float v = acc[mt][t][r] + bv;
                if (ACT == 1) v = softplus_f(v);
                C[(size_t)m * ldc + n] = v;
                if constexpr (STORE16) C16[(size_t)m * ldc16 + n] = (_Float16)v;
            }
        }
    }
}

// ---------------------------------------------------------------------------
// Depthwise conv (window 4, SAME: pad_lo=1 pad_hi=2) + SiLU for both halves.
// ---------------------------------------------------------------------------
__global__ void dwconv_silu_kernel(const float* __restrict__ xz,
                                   const float* __restrict__ Kx,
                                   const float* __restrict__ Kz,
                                   float* __restrict__ xsc,
                                   _Float16* __restrict__ xsc16,
                                   _Float16* __restrict__ ycat16) {
    const int idx = blockIdx.x * blockDim.x + threadIdx.x;
    const int c  = idx & (DHALF - 1);
    const int bl = idx >> 10;                 // DHALF == 1024
    if (bl >= MROWS) return;
    const int l = bl % LSEQ;

    float sx = 0.0f, sz = 0.0f;
#pragma unroll
    for (int w = 0; w < 4; ++w) {
        const int li = l + w - 1;
        if (li < 0 || li >= LSEQ) continue;
        const size_t rr = ((size_t)bl + (li - l)) * DINNER;
        sx += xz[rr + c]         * Kx[w * DHALF + c];
        sz += xz[rr + DHALF + c] * Kz[w * DHALF + c];
    }
    sx = silu_f(sx);
    sz = silu_f(sz);
    xsc[(size_t)bl * DHALF + c]             = sx;
    xsc16[(size_t)bl * DHALF + c]           = (_Float16)sx;
    ycat16[(size_t)bl * DINNER + DHALF + c] = (_Float16)sz;
}

// ---------------------------------------------------------------------------
// Selective scan: one lane per (b,d) channel, h[16] in registers, serial in L.
// ---------------------------------------------------------------------------
__global__ void selective_scan_kernel(const float* __restrict__ xsc,
                                      const float* __restrict__ delta,
                                      const float* __restrict__ xdbl,
                                      const float* __restrict__ A_log,
                                      const float* __restrict__ D_param,
                                      _Float16* __restrict__ ycat16) {
    const int t = blockIdx.x * blockDim.x + threadIdx.x;   // 0..8191
    const int d = t & (DHALF - 1);
    const int b = t >> 10;

    float Ad[DSTATE];
#pragma unroll
    for (int n = 0; n < DSTATE; ++n) Ad[n] = -__expf(A_log[d * DSTATE + n]);
    const float Dp = D_param[d];

    float h[DSTATE];
#pragma unroll
    for (int n = 0; n < DSTATE; ++n) h[n] = 0.0f;

    for (int l = 0; l < LSEQ; ++l) {
        const size_t row = (size_t)b * LSEQ + l;
        const float dt = delta[row * DHALF + d];
        const float u  = xsc[row * DHALF + d];

        float Bv[DSTATE], Cv[DSTATE];
        const float4* Bp = (const float4*)(xdbl + row * XDBL_N + DTRANK);
        const float4* Cp = (const float4*)(xdbl + row * XDBL_N + DTRANK + DSTATE);
#pragma unroll
        for (int q = 0; q < 4; ++q) {
            float4 bv4 = Bp[q];
            float4 cv4 = Cp[q];
            Bv[4 * q + 0] = bv4.x; Bv[4 * q + 1] = bv4.y;
            Bv[4 * q + 2] = bv4.z; Bv[4 * q + 3] = bv4.w;
            Cv[4 * q + 0] = cv4.x; Cv[4 * q + 1] = cv4.y;
            Cv[4 * q + 2] = cv4.z; Cv[4 * q + 3] = cv4.w;
        }

        const float dtu = dt * u;
        float y = 0.0f;
#pragma unroll
        for (int n = 0; n < DSTATE; ++n) {
            const float dA = __expf(dt * Ad[n]);
            h[n] = dA * h[n] + dtu * Bv[n];
            y += h[n] * Cv[n];
        }
        ycat16[row * DINNER + d] = (_Float16)(y + u * Dp);
    }
}

// ---------------------------------------------------------------------------
extern "C" void kernel_launch(void* const* d_in, const int* in_sizes, int n_in,
                              void* d_out, int out_size, void* d_ws, size_t ws_size,
                              hipStream_t stream) {
    const float* x      = (const float*)d_in[0];
    const float* W_in   = (const float*)d_in[1];
    const float* K_x    = (const float*)d_in[2];
    const float* K_z    = (const float*)d_in[3];
    const float* W_xp   = (const float*)d_in[4];
    const float* W_dt   = (const float*)d_in[5];
    const float* b_dt   = (const float*)d_in[6];
    const float* A_log  = (const float*)d_in[7];
    const float* D_par  = (const float*)d_in[8];
    const float* W_out  = (const float*)d_in[9];
    const float* b_out  = (const float*)d_in[10];
    float* out = (float*)d_out;

    // ---- workspace layout (256B aligned regions) ----
    char* p = (char*)d_ws;
    auto alloc = [&](size_t bytes) {
        char* r = p;
        p += (bytes + 255) & ~(size_t)255;
        return r;
    };
    float*    xz      = (float*)   alloc((size_t)MROWS * DINNER * 4);
    float*    xsc     = (float*)   alloc((size_t)MROWS * DHALF  * 4);
    float*    delta   = (float*)   alloc((size_t)MROWS * DHALF  * 4);
    float*    xdbl    = (float*)   alloc((size_t)MROWS * XDBL_N * 4);
    _Float16* x16     = (_Float16*)alloc((size_t)MROWS * DMODEL * 2);
    _Float16* xsc16   = (_Float16*)alloc((size_t)MROWS * DHALF  * 2);
    _Float16* xdbl16  = (_Float16*)alloc((size_t)MROWS * XDBL_N * 2);
    _Float16* ycat16  = (_Float16*)alloc((size_t)MROWS * DINNER * 2);
    _Float16* WTin16  = (_Float16*)alloc((size_t)DMODEL * DINNER * 2);  // [N=2048][K=1024]
    _Float16* WTxp16  = (_Float16*)alloc((size_t)DHALF * XDBL_N * 2);   // [N=96][K=1024]
    _Float16* WTdt16  = (_Float16*)alloc((size_t)DTRANK * DHALF * 2);   // [N=1024][K=64]
    _Float16* WTout16 = (_Float16*)alloc((size_t)DINNER * DMODEL * 2);  // [N=1024][K=2048]

    const dim3 blk(128);

    // ---- 0) precision prep: convert x and transpose-convert weights ----
    cvt16_kernel<<<dim3((MROWS * DMODEL) / 256), dim3(256), 0, stream>>>(
        x, x16, MROWS * DMODEL);
    cvtT16_kernel<<<dim3((DMODEL * DINNER) / 256), dim3(256), 0, stream>>>(
        W_in, WTin16, DMODEL, DINNER);
    cvtT16_kernel<<<dim3((DHALF * XDBL_N) / 256), dim3(256), 0, stream>>>(
        W_xp, WTxp16, DHALF, XDBL_N);
    cvtT16_kernel<<<dim3((DTRANK * DHALF) / 256), dim3(256), 0, stream>>>(
        W_dt, WTdt16, DTRANK, DHALF);
    cvtT16_kernel<<<dim3((DINNER * DMODEL) / 256), dim3(256), 0, stream>>>(
        W_out, WTout16, DINNER, DMODEL);

    // ---- 1) xz = x @ W_in            M=16384 N=2048 K=1024 ----
    wmma_gemm16_kernel<0, false, false><<<dim3(MROWS / 128, DINNER / 64), blk, 0, stream>>>(
        x16, WTin16, xz, nullptr, nullptr,
        MROWS, DINNER, DMODEL, DMODEL, DINNER, 0);

    // ---- 2) depthwise conv + SiLU ----
    dwconv_silu_kernel<<<dim3((MROWS * DHALF) / 256), dim3(256), 0, stream>>>(
        xz, K_x, K_z, xsc, xsc16, ycat16);

    // ---- 3) x_dbl = xs_c @ W_xp      M=16384 N=96 K=1024 (guarded tiles) ----
    wmma_gemm16_kernel<0, true, true><<<dim3(MROWS / 128, (XDBL_N + 63) / 64), blk, 0, stream>>>(
        xsc16, WTxp16, xdbl, xdbl16, nullptr,
        MROWS, XDBL_N, DHALF, DHALF, XDBL_N, XDBL_N);

    // ---- 4) delta = softplus(dt_low @ W_dt + b_dt)   M=16384 N=1024 K=64 ----
    wmma_gemm16_kernel<1, false, false><<<dim3(MROWS / 128, DHALF / 64), blk, 0, stream>>>(
        xdbl16, WTdt16, delta, nullptr, b_dt,
        MROWS, DHALF, DTRANK, XDBL_N, DHALF, 0);

    // ---- 5) selective scan -> left half of ycat16 ----
    selective_scan_kernel<<<dim3((BSZ * DHALF) / 256), dim3(256), 0, stream>>>(
        xsc, delta, xdbl, A_log, D_par, ycat16);

    // ---- 6) out = ycat @ W_out + b_out   M=16384 N=1024 K=2048 ----
    wmma_gemm16_kernel<0, false, false><<<dim3(MROWS / 128, DMODEL / 64), blk, 0, stream>>>(
        ycat16, WTout16, out, nullptr, b_out,
        MROWS, DMODEL, DINNER, DINNER, DMODEL, 0);
}
